// EfficientCrossScanSS2D_42228118454704
// MI455X (gfx1250) — compile-verified, matrix operand order checked by reading, not run on hardware
//
#include <hip/hip_runtime.h>
#include <cstdint>
#include <cstddef>

// ---------------- problem constants ----------------
#define B_N    2
#define C_N    128
#define H_N    120
#define W_N    120
#define L_N    (H_N * W_N)        // 14400
#define NROW   (B_N * L_N)        // 28800
#define D_IN   160
#define D_ST   24
#define DT_R   8
#define NXP    64                 // xdbc padded width (56 -> 64)
#define NCH    120                // chunks per batch for the scan
#define CHL    (L_N / NCH)        // 120 steps per chunk

#define GT_ROWS 128               // GEMM block: 8 waves * 16 rows
#define GT_COLS 64                // GEMM block: 4 col tiles of 16
#define K_MAX   160

typedef __attribute__((ext_vector_type(16))) _Float16 v16h;
typedef __attribute__((ext_vector_type(8)))  float    v8f;

union Frag16 { v16h v; unsigned u[8]; };

// Load a 16x32 f16 WMMA fragment from a row-major matrix (ld = K).
// rowBase = first row of the 16-row tile, kb = K offset (multiple of 32).
// CDNA5 layout: lanes 0-15 hold K pairs {0..3, 8..11}, lanes 16-31 hold {4..7, 12..15}.
// Templated on pointer type so LDS-resident panels keep addrspace(3) after inlining.
template <typename PT>
__device__ inline Frag16 load_frag(PT p, int rowBase, int ld, int kb, int lane) {
  Frag16 f;
  int r = rowBase + (lane & 15);
  int half = lane >> 4;
  auto* src = reinterpret_cast<const unsigned*>(&p[(size_t)r * ld + kb]);
#pragma unroll
  for (int j = 0; j < 8; ++j) {
    int pj = (j < 4) ? (j + half * 4) : (j + 4 + half * 4);
    f.u[j] = src[pj];
  }
  return f;
}

// ---------------- weight conversion ----------------
__global__ void cvt_f16_kernel(const float* __restrict__ s, _Float16* __restrict__ d, int n) {
  int i = blockIdx.x * 256 + threadIdx.x;
  if (i < n) d[i] = (_Float16)s[i];
}

__global__ void cvt_wx_pad_kernel(const float* __restrict__ s, _Float16* __restrict__ d) {
  int i = blockIdx.x * 256 + threadIdx.x;           // 64 x 160
  if (i >= NXP * D_IN) return;
  int n = i / D_IN, k = i % D_IN;
  d[i] = (n < 56) ? (_Float16)s[n * D_IN + k] : (_Float16)0.f;
}

// ---------------- cross-scan + layernorm -> seq f16 ----------------
__global__ void crossscan_ln_kernel(const float* __restrict__ x,
                                    const float* __restrict__ nw,
                                    const float* __restrict__ nb,
                                    _Float16* __restrict__ seqh) {
  int l = blockIdx.x, b = blockIdx.y, c = threadIdx.x;   // c in [0,128)
  int grp = c >> 5;
  int src;
  if (grp == 0)      src = l;
  else if (grp == 1) src = L_N - 1 - l;
  else if (grp == 2) { int w = l / H_N, h = l % H_N; src = h * W_N + w; }
  else { int lp = L_N - 1 - l; int w = lp / H_N, h = lp % H_N; src = h * W_N + w; }
  float v = x[((size_t)b * C_N + c) * L_N + src];

  __shared__ float sA[128], sB[128];
  sA[c] = v; sB[c] = v * v;
  __syncthreads();
  for (int off = 64; off > 0; off >>= 1) {
    if (c < off) { sA[c] += sA[c + off]; sB[c] += sB[c + off]; }
    __syncthreads();
  }
  float mu  = sA[0] * (1.f / 128.f);
  float var = sB[0] * (1.f / 128.f) - mu * mu;
  float y = (v - mu) * rsqrtf(var + 1e-5f) * nw[c] + nb[c];
  seqh[((size_t)b * L_N + l) * C_N + c] = (_Float16)y;
}

// ---------------- WMMA GEMM: C(MxN) = A(MxK) * Bw(NxK)^T ----------------
// B weight panel (64 x K) staged in LDS once per block; each wave computes 16x64.
// mode 0: f32 store (ld=ldo). mode 1: f16 store. mode 2: fused residual epilogue.
__global__ __launch_bounds__(256) void
wmma_gemm_kernel(const _Float16* __restrict__ A,
                 const _Float16* __restrict__ Bw,
                 int M, int N, int K, int mode,
                 float* __restrict__ outf,
                 _Float16* __restrict__ outh, int ldo,
                 const float* __restrict__ resid,
                 const float* __restrict__ scale) {
  __shared__ _Float16 Bs[GT_COLS * K_MAX];
  int tid = threadIdx.x;
  int col0 = blockIdx.y * GT_COLS;

  // cooperative stage of the contiguous B panel (rows col0..col0+63, ld = K)
  {
    const unsigned* bsrc = reinterpret_cast<const unsigned*>(Bw + (size_t)col0 * K);
    unsigned* bdst = reinterpret_cast<unsigned*>(Bs);
    int npair = (GT_COLS * K) >> 1;
    for (int i = tid; i < npair; i += 256) bdst[i] = bsrc[i];
  }
  __syncthreads();

  int wave = tid >> 5, lane = tid & 31, half = lane >> 4;
  int m0 = (blockIdx.x * 8 + wave) * 16;
  if (m0 >= M) return;

  v8f acc[4];
#pragma unroll
  for (int t = 0; t < 4; ++t) acc[t] = (v8f){0, 0, 0, 0, 0, 0, 0, 0};

  for (int kb = 0; kb < K; kb += 32) {
    Frag16 a = load_frag(A, m0, K, kb, lane);
    if (kb + 32 < K)   // prefetch next K-slice of the streamed A operand
      __builtin_prefetch(A + (size_t)(m0 + (lane & 15)) * K + kb + 32, 0, 1);
#pragma unroll
    for (int t = 0; t < 4; ++t) {
      Frag16 b = load_frag(Bs, t * 16, K, kb, lane);
      acc[t] = __builtin_amdgcn_wmma_f32_16x16x32_f16(false, a.v, false, b.v,
                                                      (short)0, acc[t], false, false);
    }
  }

#pragma unroll
  for (int t = 0; t < 4; ++t) {
    int n = col0 + t * 16 + (lane & 15);
#pragma unroll
    for (int r = 0; r < 8; ++r) {
      int m = m0 + r + half * 8;
      if (mode == 0) {
        outf[(size_t)m * ldo + n] = acc[t][r];
      } else if (mode == 1) {
        outh[(size_t)m * ldo + n] = (_Float16)acc[t][r];
      } else {
        int bb = m / L_N, hw = m % L_N;
        size_t i0 = ((size_t)bb * C_N + n) * L_N + hw;
        outf[i0] = resid[i0] + scale[0] * acc[t][r];
      }
    }
  }
}

// ---------------- depthwise causal conv (K=4) + SiLU ----------------
__global__ void conv_silu_kernel(const float* __restrict__ xz,
                                 const float* __restrict__ cw,
                                 const float* __restrict__ cb,
                                 float* __restrict__ u, _Float16* __restrict__ uh) {
  int b = blockIdx.x / NCH, ch = blockIdx.x % NCH;
  int d = threadIdx.x;                       // 160 threads
  int l0 = ch * CHL;
  float w0 = cw[d * 4 + 0], w1 = cw[d * 4 + 1], w2 = cw[d * 4 + 2], w3 = cw[d * 4 + 3];
  float bias = cb[d];
  auto ldx = [&](int l) -> float {
    return (l < 0) ? 0.f : xz[((size_t)b * L_N + l) * 320 + d];
  };
  float xm1 = ldx(l0 - 1), xm2 = ldx(l0 - 2), xm3 = ldx(l0 - 3);
  for (int t = 0; t < CHL; ++t) {
    int l = l0 + t;
    float x0 = xz[((size_t)b * L_N + l) * 320 + d];
    float a = w3 * x0 + w2 * xm1 + w1 * xm2 + w0 * xm3 + bias;
    float s = a / (1.f + __expf(-a));        // SiLU
    size_t row = (size_t)b * L_N + l;
    u[row * D_IN + d]  = s;
    uh[row * D_IN + d] = (_Float16)s;
    xm3 = xm2; xm2 = xm1; xm1 = x0;
  }
}

// ---------------- delta = softplus(dt @ W_dt^T + b_dt) ----------------
__global__ void delta_kernel(const float* __restrict__ xdbc,
                             const float* __restrict__ Wdt,
                             const float* __restrict__ bdt,
                             float* __restrict__ delta) {
  int row = blockIdx.x, d = threadIdx.x;     // 160 threads
  const float* dt = xdbc + (size_t)row * NXP;
  float acc = bdt[d];
#pragma unroll
  for (int r = 0; r < DT_R; ++r) acc += dt[r] * Wdt[d * DT_R + r];
  delta[(size_t)row * D_IN + d] = (acc > 20.f) ? acc : logf(1.f + __expf(acc));
}

// ---------------- chunked selective scan ----------------
__global__ void scan_phase1(const float* __restrict__ delta, const float* __restrict__ u,
                            const float* __restrict__ xdbc, const float* __restrict__ Alog,
                            float* __restrict__ aprod, float* __restrict__ bacc) {
  __shared__ float lB[CHL * D_ST];
  int b = blockIdx.x / NCH, ch = blockIdx.x % NCH;
  int d = threadIdx.x;                       // 160 threads
  int row0 = b * L_N + ch * CHL;
  for (int i = d; i < CHL * D_ST; i += D_IN) {
    int t = i / D_ST, n = i % D_ST;
    lB[i] = xdbc[(size_t)(row0 + t) * NXP + DT_R + n];
  }
  __syncthreads();
  float Ar[D_ST], ap[D_ST], bc[D_ST];
#pragma unroll
  for (int n = 0; n < D_ST; ++n) {
    Ar[n] = -__expf(Alog[d * D_ST + n]); ap[n] = 1.f; bc[n] = 0.f;
  }
  for (int t = 0; t < CHL; ++t) {
    size_t row = row0 + t;
    float dl  = delta[row * D_IN + d];
    float dbu = dl * u[row * D_IN + d];
    const float* Bt = &lB[t * D_ST];
#pragma unroll
    for (int n = 0; n < D_ST; ++n) {
      float an = __expf(dl * Ar[n]);
      bc[n] = an * bc[n] + dbu * Bt[n];
      ap[n] *= an;
    }
  }
  size_t o = (((size_t)b * NCH + ch) * D_IN + d) * D_ST;
#pragma unroll
  for (int n = 0; n < D_ST; ++n) { aprod[o + n] = ap[n]; bacc[o + n] = bc[n]; }
}

__global__ void scan_phase2(const float* __restrict__ aprod, const float* __restrict__ bacc,
                            float* __restrict__ hinit) {
  int tid = blockIdx.x * 256 + threadIdx.x;
  if (tid >= B_N * D_IN * D_ST) return;
  int b = tid / (D_IN * D_ST);
  int rem = tid % (D_IN * D_ST);
  int d = rem / D_ST, n = rem % D_ST;
  float h = 0.f;
  for (int c = 0; c < NCH; ++c) {
    size_t idx = (((size_t)b * NCH + c) * D_IN + d) * D_ST + n;
    hinit[idx] = h;
    h = aprod[idx] * h + bacc[idx];
  }
}

__global__ void scan_phase3(const float* __restrict__ delta, const float* __restrict__ u,
                            const float* __restrict__ xdbc, const float* __restrict__ Alog,
                            const float* __restrict__ hinit, const float* __restrict__ Dv,
                            const float* __restrict__ xz, _Float16* __restrict__ ygate) {
  __shared__ float lB[CHL * D_ST];
  __shared__ float lC[CHL * D_ST];
  int b = blockIdx.x / NCH, ch = blockIdx.x % NCH;
  int d = threadIdx.x;
  int row0 = b * L_N + ch * CHL;
  for (int i = d; i < CHL * D_ST; i += D_IN) {
    int t = i / D_ST, n = i % D_ST;
    lB[i] = xdbc[(size_t)(row0 + t) * NXP + DT_R + n];
    lC[i] = xdbc[(size_t)(row0 + t) * NXP + DT_R + D_ST + n];
  }
  __syncthreads();
  float Ar[D_ST], h[D_ST];
  size_t hid = (((size_t)b * NCH + ch) * D_IN + d) * D_ST;
#pragma unroll
  for (int n = 0; n < D_ST; ++n) {
    Ar[n] = -__expf(Alog[d * D_ST + n]);
    h[n]  = hinit[hid + n];
  }
  float Dd = Dv[d];
  for (int t = 0; t < CHL; ++t) {
    size_t row = row0 + t;
    float dl  = delta[row * D_IN + d];
    float uu  = u[row * D_IN + d];
    float dbu = dl * uu;
    const float* Bt = &lB[t * D_ST];
    const float* Ct = &lC[t * D_ST];
    float y = 0.f;
#pragma unroll
    for (int n = 0; n < D_ST; ++n) {
      float an = __expf(dl * Ar[n]);
      h[n] = an * h[n] + dbu * Bt[n];
      y += h[n] * Ct[n];
    }
    float yo = y + uu * Dd;
    float z  = xz[row * 320 + D_IN + d];
    float sz = z / (1.f + __expf(-z));
    ygate[row * D_IN + d] = (_Float16)(yo * sz);
  }
}

// ---------------- cross-merge: out_seq(B,L,C) -> spatial^T rows=(b,hw) ----------------
__global__ void crossmerge_kernel(const _Float16* __restrict__ oseq,
                                  _Float16* __restrict__ spat) {
  int idx = blockIdx.x * 256 + threadIdx.x;      // NROW*128 total
  if (idx >= NROW * C_N) return;
  int row = idx >> 7, c = idx & 127;
  int b = row / L_N, hw = row % L_N;
  int grp = c >> 5, sl;
  if (grp == 0)      sl = hw;
  else if (grp == 1) sl = L_N - 1 - hw;
  else {
    int h = hw / W_N, w = hw % W_N;
    int t = w * H_N + h;
    sl = (grp == 2) ? t : (L_N - 1 - t);
  }
  spat[idx] = oseq[((size_t)b * L_N + sl) * C_N + c];
}

// ---------------- launcher ----------------
static inline size_t alignup(size_t x) { return (x + 255) & ~(size_t)255; }

extern "C" void kernel_launch(void* const* d_in, const int* in_sizes, int n_in,
                              void* d_out, int out_size, void* d_ws, size_t ws_size,
                              hipStream_t stream) {
  const float* x      = (const float*)d_in[0];
  const float* norm_w = (const float*)d_in[1];
  const float* norm_b = (const float*)d_in[2];
  const float* W_in   = (const float*)d_in[3];   // (320,128)
  const float* conv_w = (const float*)d_in[4];   // (160,4)
  const float* conv_b = (const float*)d_in[5];
  const float* W_x    = (const float*)d_in[6];   // (56,160)
  const float* W_dt   = (const float*)d_in[7];   // (160,8)
  const float* b_dt   = (const float*)d_in[8];
  const float* A_log  = (const float*)d_in[9];   // (160,24)
  const float* Dv     = (const float*)d_in[10];
  const float* W_out  = (const float*)d_in[11];  // (128,160)
  const float* Wfu    = (const float*)d_in[12];  // (128,128)
  const float* scale  = (const float*)d_in[13];
  float* out = (float*)d_out;

  // workspace carve-up
  char* ws = (char*)d_ws;
  size_t off = 0;
  auto take = [&](size_t bytes) { char* p = ws + off; off += alignup(bytes); return p; };
  _Float16* seq_h  = (_Float16*)take((size_t)NROW * C_N * 2);
  float*    xz     = (float*)   take((size_t)NROW * 320 * 4);
  float*    u      = (float*)   take((size_t)NROW * D_IN * 4);
  _Float16* u_h    = (_Float16*)take((size_t)NROW * D_IN * 2);
  float*    xdbc   = (float*)   take((size_t)NROW * NXP * 4);
  float*    delta  = (float*)   take((size_t)NROW * D_IN * 4);
  float*    aprod  = (float*)   take((size_t)B_N * NCH * D_IN * D_ST * 4);
  float*    bacc   = (float*)   take((size_t)B_N * NCH * D_IN * D_ST * 4);
  float*    hinit  = (float*)   take((size_t)B_N * NCH * D_IN * D_ST * 4);
  _Float16* ygate  = (_Float16*)take((size_t)NROW * D_IN * 2);
  _Float16* oseq_h = (_Float16*)take((size_t)NROW * C_N * 2);
  _Float16* spat_h = (_Float16*)take((size_t)NROW * C_N * 2);
  _Float16* Win_h  = (_Float16*)take((size_t)320 * 128 * 2);
  _Float16* Wx_h   = (_Float16*)take((size_t)NXP * D_IN * 2);
  _Float16* Wout_h = (_Float16*)take((size_t)128 * D_IN * 2);
  _Float16* Wfu_h  = (_Float16*)take((size_t)128 * 128 * 2);
  (void)ws_size; (void)n_in; (void)in_sizes; (void)out_size;

  // 1) weight conversion
  cvt_f16_kernel<<<(320 * 128 + 255) / 256, 256, 0, stream>>>(W_in, Win_h, 320 * 128);
  cvt_wx_pad_kernel<<<(NXP * D_IN + 255) / 256, 256, 0, stream>>>(W_x, Wx_h);
  cvt_f16_kernel<<<(128 * D_IN + 255) / 256, 256, 0, stream>>>(W_out, Wout_h, 128 * D_IN);
  cvt_f16_kernel<<<(128 * 128 + 255) / 256, 256, 0, stream>>>(Wfu, Wfu_h, 128 * 128);

  // 2) cross-scan + layernorm
  crossscan_ln_kernel<<<dim3(L_N, B_N), 128, 0, stream>>>(x, norm_w, norm_b, seq_h);

  // 3) GEMM1: xz = seq @ W_in^T   (M=28800, N=320, K=128)
  wmma_gemm_kernel<<<dim3(NROW / GT_ROWS, 320 / GT_COLS), 256, 0, stream>>>(
      seq_h, Win_h, NROW, 320, 128, 0, xz, nullptr, 320, nullptr, nullptr);

  // 4) depthwise causal conv + SiLU
  conv_silu_kernel<<<B_N * NCH, D_IN, 0, stream>>>(xz, conv_w, conv_b, u, u_h);

  // 5) GEMM2: xdbc = u @ W_x^T (padded N=64, K=160)
  wmma_gemm_kernel<<<dim3(NROW / GT_ROWS, NXP / GT_COLS), 256, 0, stream>>>(
      u_h, Wx_h, NROW, NXP, D_IN, 0, xdbc, nullptr, NXP, nullptr, nullptr);

  // 6) delta
  delta_kernel<<<NROW, D_IN, 0, stream>>>(xdbc, W_dt, b_dt, delta);

  // 7) chunked selective scan
  scan_phase1<<<B_N * NCH, D_IN, 0, stream>>>(delta, u, xdbc, A_log, aprod, bacc);
  scan_phase2<<<(B_N * D_IN * D_ST + 255) / 256, 256, 0, stream>>>(aprod, bacc, hinit);
  scan_phase3<<<B_N * NCH, D_IN, 0, stream>>>(delta, u, xdbc, A_log, hinit, Dv, xz, ygate);

  // 8) GEMM3: out_seq = ygate @ W_out^T (N=128, K=160), f16 out
  wmma_gemm_kernel<<<dim3(NROW / GT_ROWS, 128 / GT_COLS), 256, 0, stream>>>(
      ygate, Wout_h, NROW, 128, D_IN, 1, nullptr, oseq_h, 128, nullptr, nullptr);

  // 9) cross-merge permutation
  crossmerge_kernel<<<(NROW * C_N + 255) / 256, 256, 0, stream>>>(oseq_h, spat_h);

  // 10) GEMM4: fusion + residual epilogue -> d_out (N=128, K=128)
  wmma_gemm_kernel<<<dim3(NROW / GT_ROWS, 128 / GT_COLS), 256, 0, stream>>>(
      spat_h, Wfu_h, NROW, 128, 128, 2, out, nullptr, 0, x, scale);
}
